// ElementwiseMLP_7937099563198
// MI455X (gfx1250) — matrix operand order repaired and checked by reading, not compiled, not run on hardware
//
#include <hip/hip_runtime.h>
#include <hip/hip_bf16.h>

typedef __attribute__((ext_vector_type(16))) _Float16 v16h;
typedef __attribute__((ext_vector_type(8)))  _Float16 v8h;
typedef __attribute__((ext_vector_type(8)))  float    v8f;

#define N_ATOMS 200000
#define N_IN    256
#define N_HID   128
#define N_ELEM  5
#define TM      64                                  // atoms per block
#define TILES_PER_E ((N_ATOMS + TM - 1) / TM)       // 3125

__global__ void init_cnt(int* cnt) {
    if (threadIdx.x < 8) cnt[threadIdx.x] = 0;
}

__global__ void bucket(const int* __restrict__ z, int* __restrict__ cnt,
                       int* __restrict__ lists) {
    int i = blockIdx.x * blockDim.x + threadIdx.x;
    if (i >= N_ATOMS) return;
    int a = z[i];
    int e = (a == 1) ? 0 : (a == 6) ? 1 : (a == 7) ? 2 : (a == 8) ? 3 : 4;
    int p = atomicAdd(&cnt[e], 1);
    lists[e * N_ATOMS + p] = i;
}

#define WMMA16(A, B, C) \
    __builtin_amdgcn_wmma_f32_16x16x32_f16(false, (A), false, (B), (short)0, (C), false, false)

__global__ __launch_bounds__(128) void moe_mlp(
    const float* __restrict__ rep, const float* __restrict__ W1,
    const float* __restrict__ b1,  const float* __restrict__ W2,
    const float* __restrict__ b2,  const int* __restrict__ cnt,
    const int* __restrict__ lists, float* __restrict__ out) {

    __shared__ __align__(32) _Float16 Xs[TM * N_IN];   // 32 KB : 64 gathered atoms
    __shared__ __align__(32) _Float16 Ws[64 * N_IN];   // 32 KB : half of W1[e]
    __shared__ float pacc[8][TM];                      // fixed-slot partials (2 KB)
    __shared__ int   aidx[TM];

    const int e  = blockIdx.x / TILES_PER_E;
    const int m0 = (blockIdx.x % TILES_PER_E) * TM;
    const int ce = cnt[e];
    if (m0 >= ce) return;                              // uniform early exit
    const int nA = (ce - m0 < TM) ? (ce - m0) : TM;

    const int tid  = threadIdx.x;
    const int lane = tid & 31;
    const int wave = tid >> 5;
    const int lm   = lane & 15;   // M / N striping within half-wave
    const int lh   = lane >> 4;   // which half of the wave

    // 2x2 register block per wave: msub {mb2, mb2+1}, ntile {nb2, nb2+1}
    const int mb2 = (wave & 1) * 2;
    const int nb2 = (wave >> 1) * 2;

    if (tid < TM) aidx[tid] = (tid < nA) ? lists[e * N_ATOMS + m0 + tid] : 0;
    __syncthreads();

    // Stage gathered X tile (64 x 256) as f16; 2 contiguous 1KB rows / iter.
    const float4* rep4 = (const float4*)rep;
    for (int i = tid; i < TM * (N_IN / 4); i += 128) {
        int r = i >> 6, c4 = i & 63;
        float4 v = make_float4(0.f, 0.f, 0.f, 0.f);
        if (r < nA) v = rep4[(size_t)aidx[r] * (N_IN / 4) + c4];
        _Float16* d = &Xs[r * N_IN + c4 * 4];
        d[0] = (_Float16)v.x; d[1] = (_Float16)v.y;
        d[2] = (_Float16)v.z; d[3] = (_Float16)v.w;
    }

    for (int phase = 0; phase < 2; ++phase) {
        __syncthreads();
        // Stage 64 hidden rows of W1[e] as f16.
        const float4* W14 =
            (const float4*)(W1 + (size_t)(e * N_HID + phase * 64) * N_IN);
        for (int i = tid; i < 64 * (N_IN / 4); i += 128) {
            int r = i >> 6, c4 = i & 63;
            float4 v = W14[r * (N_IN / 4) + c4];
            _Float16* d = &Ws[r * N_IN + c4 * 4];
            d[0] = (_Float16)v.x; d[1] = (_Float16)v.y;
            d[2] = (_Float16)v.z; d[3] = (_Float16)v.w;
        }
        __syncthreads();

        v8f acc00 = {}, acc01 = {}, acc10 = {}, acc11 = {};
        const _Float16* xr0 = &Xs[(mb2 * 16 + 16 * 0 + lm) * N_IN];
        const _Float16* xr1 = &Xs[((mb2 + 1) * 16 + lm) * N_IN];
        const _Float16* wr0 = &Ws[(nb2 * 16 + lm) * N_IN];
        const _Float16* wr1 = &Ws[((nb2 + 1) * 16 + lm) * N_IN];

        #pragma unroll
        for (int kc = 0; kc < 8; ++kc) {
            v16h a0, a1, fb0, fb1;
            *(v8h*)&a0       = *(const v8h*)(xr0 + kc * 32 + lh * 8);
            *((v8h*)&a0 + 1) = *(const v8h*)(xr0 + kc * 32 + 16 + lh * 8);
            *(v8h*)&a1       = *(const v8h*)(xr1 + kc * 32 + lh * 8);
            *((v8h*)&a1 + 1) = *(const v8h*)(xr1 + kc * 32 + 16 + lh * 8);
            *(v8h*)&fb0       = *(const v8h*)(wr0 + kc * 32 + lh * 8);
            *((v8h*)&fb0 + 1) = *(const v8h*)(wr0 + kc * 32 + 16 + lh * 8);
            *(v8h*)&fb1       = *(const v8h*)(wr1 + kc * 32 + lh * 8);
            *((v8h*)&fb1 + 1) = *(const v8h*)(wr1 + kc * 32 + 16 + lh * 8);
            // 4 independent accumulation chains (hides WMMA->WMMA hazards)
            acc00 = WMMA16(a0, fb0, acc00);
            acc10 = WMMA16(a1, fb0, acc10);
            acc01 = WMMA16(a0, fb1, acc01);
            acc11 = WMMA16(a1, fb1, acc11);
        }

        // Epilogue: bias + silu, scale by W2 column, reduce over N (lanes).
        #pragma unroll
        for (int jn = 0; jn < 2; ++jn) {
            int nt = nb2 + jn;
            int ng = phase * 64 + nt * 16 + lm;       // hidden-unit index
            float bb = b1[e * N_HID + ng];
            float w2 = W2[e * N_HID + ng];
            #pragma unroll
            for (int im = 0; im < 2; ++im) {
                int msub = mb2 + im;
                const v8f& acc = (im == 0) ? (jn == 0 ? acc00 : acc01)
                                           : (jn == 0 ? acc10 : acc11);
                float p[8];
                #pragma unroll
                for (int r = 0; r < 8; ++r) {
                    float x = acc[r] + bb;
                    float h = x / (1.f + __expf(-x)); // silu
                    p[r] = h * w2;
                }
                #pragma unroll
                for (int r = 0; r < 8; ++r) {
                    #pragma unroll
                    for (int off = 8; off > 0; off >>= 1)
                        p[r] += __shfl_xor(p[r], off, 16);
                }
                if (lm == 0) {                         // lanes 0 and 16
                    int mb = msub * 16 + lh * 8;       // C layout: M = r + 8*lh
                    #pragma unroll
                    for (int r = 0; r < 8; ++r)
                        pacc[phase * 4 + nt][mb + r] = p[r];
                }
            }
        }
    }
    __syncthreads();
    if (tid < nA) {
        float y = b2[e];
        #pragma unroll
        for (int s = 0; s < 8; ++s) y += pacc[s][tid];
        out[aidx[tid]] = y;                            // scatter to original order
    }
}

extern "C" void kernel_launch(void* const* d_in, const int* in_sizes, int n_in,
                              void* d_out, int out_size, void* d_ws, size_t ws_size,
                              hipStream_t stream) {
    const int*   an  = (const int*)d_in[0];
    const float* rep = (const float*)d_in[1];
    const float* W1  = (const float*)d_in[2];
    const float* b1  = (const float*)d_in[3];
    const float* W2  = (const float*)d_in[4];
    const float* b2  = (const float*)d_in[5];
    float* out = (float*)d_out;

    int* cnt   = (int*)d_ws;        // 64 ints
    int* lists = cnt + 64;          // 5 * N_ATOMS ints (~4 MB)

    init_cnt<<<1, 64, 0, stream>>>(cnt);
    bucket<<<(N_ATOMS + 255) / 256, 256, 0, stream>>>(an, cnt, lists);
    moe_mlp<<<N_ELEM * TILES_PER_E, 128, 0, stream>>>(rep, W1, b1, W2, b2,
                                                      cnt, lists, out);
}
